// ForwardEuler_8598524527223
// MI455X (gfx1250) — matrix operand order repaired and checked by reading, not compiled, not run on hardware
//
#include <hip/hip_runtime.h>

#define B_SZ 2048
#define T_SZ 1024
#define NX   64
#define NU   32
#define H_SZ 256
#define DT_F 0.01f

#define TM       16    // batch rows per workgroup (WMMA M)
#define NWAVE    4
#define NTHREADS 128

#define SA_STRIDE 104  // 96  + 8 pad (halfs); 208B rows, 16B aligned
#define SH_STRIDE 264  // 256 + 8 pad (halfs); 528B rows, 16B aligned

typedef __attribute__((ext_vector_type(16))) _Float16 v16h;
typedef __attribute__((ext_vector_type(8)))  _Float16 v8h;
typedef __attribute__((ext_vector_type(4)))  _Float16 v4h;
typedef __attribute__((ext_vector_type(8)))  float    v8f;
typedef __attribute__((ext_vector_type(4)))  float    v4f;

__device__ __forceinline__ float fast_tanh(float x) {
#if __has_builtin(__builtin_amdgcn_tanhf)
    // CDNA5 native transcendental: v_tanh_f32 (single TRANS32 op)
    return __builtin_amdgcn_tanhf(x);
#else
    // fallback: tanh(x) = 1 - 2/(exp(2x)+1) via v_exp_f32 / v_rcp_f32
    float e = __builtin_amdgcn_exp2f(x * 2.8853900817779268f); // 2*log2(e)
    return 1.0f - 2.0f * __builtin_amdgcn_rcpf(e + 1.0f);
#endif
}

// A-fragment (16x32 f16) from row-major LDS buffer, per ISA 16-bit A layout:
// lanes 0-15: row=lane,  K = k0+{0..7, 16..23}
// lanes 16-31: row=lane-16, K = k0+{8..15, 24..31}
__device__ __forceinline__ v16h load_a_frag(const _Float16* buf, int stride,
                                            int k0, int lane) {
    const int row = lane & 15;
    const int kb  = (lane >> 4) << 3;           // 0 or 8
    const _Float16* p = buf + row * stride + k0 + kb;
    v8h lo = *(const v8h*)p;                    // K = k0+kb .. +7
    v8h hi = *(const v8h*)(p + 16);             // K = k0+kb+16 .. +23
    v16h r;
#pragma unroll
    for (int i = 0; i < 8; ++i) { r[i] = lo[i]; r[8 + i] = hi[i]; }
    return r;
}

// B-fragment (32x16 f16) from row-major f32 weight matrix (one-time f32->f16):
// lane L holds column N = n0+L%16, K = k0+8*(L/16)+{0..7, 16..23}.
__device__ __forceinline__ v16h load_b_frag(const float* __restrict__ W, int ldw,
                                            int k0, int n0, int lane) {
    const float* p = W + (size_t)(k0 + ((lane >> 4) << 3)) * ldw + n0 + (lane & 15);
    v16h r;
#pragma unroll
    for (int i = 0; i < 16; ++i) r[i] = (_Float16)p[(size_t)((i & 7) + ((i >> 3) << 4)) * ldw];
    return r;
}

__global__ __launch_bounds__(NTHREADS)
void euler_mlp_kernel(const float* __restrict__ x0,
                      const float* __restrict__ u,
                      const float* __restrict__ W1,
                      const float* __restrict__ b1,
                      const float* __restrict__ W2,
                      const float* __restrict__ b2,
                      float*       __restrict__ out) {
    __shared__ __align__(16) _Float16 sA[TM * SA_STRIDE];   // [x(64) | u(32)] f16
    __shared__ __align__(16) _Float16 sH[TM * SH_STRIDE];   // hidden f16
    __shared__ __align__(16) float    sDx[NWAVE * TM * NX]; // dx partials f32
    __shared__ __align__(16) float    sX[TM * NX];          // carried state f32

    const int tid  = threadIdx.x;
    const int lane = tid & 31;
    const int w    = tid >> 5;              // wave id 0..3
    const int m0   = blockIdx.x * TM;       // batch-row base

    // ---- one-time: weight B-fragments into registers (constant over T) ----
    v16h bW1[3][4];                         // K tiles {0,32,64} x N tiles of 64w+16j
#pragma unroll
    for (int k = 0; k < 3; ++k)
#pragma unroll
        for (int j = 0; j < 4; ++j)
            bW1[k][j] = load_b_frag(W1, H_SZ, 32 * k, 64 * w + 16 * j, lane);

    v16h bW2[2][4];                         // K tiles {64w, 64w+32} x N tiles 16j
#pragma unroll
    for (int k = 0; k < 2; ++k)
#pragma unroll
        for (int j = 0; j < 4; ++j)
            bW2[k][j] = load_b_frag(W2, NX, 64 * w + 32 * k, 16 * j, lane);

    float b1c[4];
#pragma unroll
    for (int j = 0; j < 4; ++j) b1c[j] = b1[64 * w + 16 * j + (lane & 15)];

    // update-phase mapping: thread owns 8 contiguous state elements
    const int urow = tid >> 3;              // 0..15
    const int xcol = (tid & 7) * 8;         // 0..56
    float b2c[8];
#pragma unroll
    for (int i = 0; i < 8; ++i) b2c[i] = b2[xcol + i];

    // ---- init carried state ----
    {
        const float* xp = x0 + (size_t)(m0 + urow) * NX + xcol;
#pragma unroll
        for (int i = 0; i < 8; ++i) {
            float v = xp[i];
            sX[urow * NX + xcol + i]          = v;
            sA[urow * SA_STRIDE + xcol + i]   = (_Float16)v;
        }
    }

    const int ucol4 = (tid & 7) * 4;        // u-load: 4 floats per thread

    for (int t = 0; t < T_SZ; ++t) {
        // ---- stage u_t (f32 -> f16) into sA columns [64,96) ----
        {
            const float* up = u + (size_t)t * (B_SZ * NU)
                                + (size_t)(m0 + urow) * NU + ucol4;
            v4f uv = *(const v4f*)up;
            v4h uh;
#pragma unroll
            for (int i = 0; i < 4; ++i) uh[i] = (_Float16)uv[i];
            *(v4h*)&sA[urow * SA_STRIDE + NX + ucol4] = uh;
            if (t + 1 < T_SZ) __builtin_prefetch(up + (size_t)B_SZ * NU, 0, 0);
        }
        __syncthreads();

        // ---- GEMM1: h = tanh([x,u] @ W1 + b1), wave w owns N in [64w,64w+64) ----
        {
            v16h a0 = load_a_frag(sA, SA_STRIDE, 0,  lane);
            v16h a1 = load_a_frag(sA, SA_STRIDE, 32, lane);
            v16h a2 = load_a_frag(sA, SA_STRIDE, 64, lane);
            const int ncl  = lane & 15;
            const int mrow = (lane >> 4) * 8;
#pragma unroll
            for (int j = 0; j < 4; ++j) {
                v8f c;
#pragma unroll
                for (int r = 0; r < 8; ++r) c[r] = b1c[j];   // bias (per-N)
                c = __builtin_amdgcn_wmma_f32_16x16x32_f16(false, a0, false, bW1[0][j], (short)0, c, false, false);
                c = __builtin_amdgcn_wmma_f32_16x16x32_f16(false, a1, false, bW1[1][j], (short)0, c, false, false);
                c = __builtin_amdgcn_wmma_f32_16x16x32_f16(false, a2, false, bW1[2][j], (short)0, c, false, false);
                const int ncol = 64 * w + 16 * j + ncl;
                // tanh all 8 first (independent TRANS32 chains pipeline), then store
                float th[8];
#pragma unroll
                for (int r = 0; r < 8; ++r) th[r] = fast_tanh(c[r]);
#pragma unroll
                for (int r = 0; r < 8; ++r)
                    sH[(mrow + r) * SH_STRIDE + ncol] = (_Float16)th[r];
            }
        }
        __syncthreads();

        // ---- GEMM2: dx_partial = h[:,64w:64w+64] @ W2[64w:64w+64,:] ----
        {
            v16h h0 = load_a_frag(sH, SH_STRIDE, 64 * w,      lane);
            v16h h1 = load_a_frag(sH, SH_STRIDE, 64 * w + 32, lane);
            const int ncl  = lane & 15;
            const int mrow = (lane >> 4) * 8;
            float* dxp = sDx + w * (TM * NX);
#pragma unroll
            for (int j = 0; j < 4; ++j) {
                v8f c;
#pragma unroll
                for (int r = 0; r < 8; ++r) c[r] = 0.0f;
                c = __builtin_amdgcn_wmma_f32_16x16x32_f16(false, h0, false, bW2[0][j], (short)0, c, false, false);
                c = __builtin_amdgcn_wmma_f32_16x16x32_f16(false, h1, false, bW2[1][j], (short)0, c, false, false);
                const int ncol = 16 * j + ncl;
#pragma unroll
                for (int r = 0; r < 8; ++r)
                    dxp[(mrow + r) * NX + ncol] = c[r];
            }
        }
        __syncthreads();

        // ---- reduce partials, Euler update, emit xhat[t] ----
        {
            float acc[8];
#pragma unroll
            for (int i = 0; i < 8; ++i) acc[i] = b2c[i];
            const int off = urow * NX + xcol;
#pragma unroll
            for (int wv = 0; wv < NWAVE; ++wv) {
                const float* p = sDx + wv * (TM * NX) + off;
#pragma unroll
                for (int i = 0; i < 8; ++i) acc[i] += p[i];
            }
            float xn[8];
#pragma unroll
            for (int i = 0; i < 8; ++i) {
                float v = sX[off + i] + DT_F * acc[i];
                xn[i] = v;
                sX[off + i] = v;
                sA[urow * SA_STRIDE + xcol + i] = (_Float16)v;
            }
            float* op = out + (size_t)t * (B_SZ * NX)
                            + (size_t)(m0 + urow) * NX + xcol;
            *(v4f*)op       = (v4f){xn[0], xn[1], xn[2], xn[3]};
            *((v4f*)op + 1) = (v4f){xn[4], xn[5], xn[6], xn[7]};
        }
        __syncthreads();
    }
}

extern "C" void kernel_launch(void* const* d_in, const int* in_sizes, int n_in,
                              void* d_out, int out_size, void* d_ws, size_t ws_size,
                              hipStream_t stream) {
    (void)in_sizes; (void)n_in; (void)out_size; (void)d_ws; (void)ws_size;
    const float* x0 = (const float*)d_in[0];
    const float* u  = (const float*)d_in[1];
    const float* W1 = (const float*)d_in[2];
    const float* b1 = (const float*)d_in[3];
    const float* W2 = (const float*)d_in[4];
    const float* b2 = (const float*)d_in[5];
    float* out = (float*)d_out;

    euler_mlp_kernel<<<dim3(B_SZ / TM), dim3(NTHREADS), 0, stream>>>(
        x0, u, W1, b1, W2, b2, out);
}
